// _PointnetSAModuleBase_57793079935582
// MI455X (gfx1250) — matrix operand order repaired and checked by reading, not compiled, not run on hardware
//
#include <hip/hip_runtime.h>
#include <hip/hip_bf16.h>

#define Bb 4
#define Nn 16384
#define Cc 64
#define Ss 1024
#define NSs 32
#define RADIUS 0.2f
#define K0P 96          // layer-0 K (64 feat + 3 xyz) padded to 3 K-tiles of 32
#define H3 128

typedef __attribute__((ext_vector_type(16))) _Float16 v16h;
typedef __attribute__((ext_vector_type(8)))  _Float16 v8h;
typedef __attribute__((ext_vector_type(8)))  float    v8f;

// ---------------------------------------------------------------------------
// Kernel 1: furthest point sampling. One block per batch, 1024 threads,
// 16 points per thread held in registers. Writes new_xyz (B,S,3) directly.
// ---------------------------------------------------------------------------
__global__ __launch_bounds__(1024) void fps_kernel(const float* __restrict__ xyz,
                                                   float* __restrict__ new_xyz) {
    const int b   = blockIdx.x;
    const int tid = threadIdx.x;
    const float* x = xyz + (size_t)b * Nn * 3;

    float px[16], py[16], pz[16], dist[16];
#pragma unroll
    for (int i = 0; i < 16; ++i) {
        int p = i * 1024 + tid;
        px[i] = x[p * 3 + 0];
        py[i] = x[p * 3 + 1];
        pz[i] = x[p * 3 + 2];
        dist[i] = 3.0e38f;
    }

    __shared__ float s_cx, s_cy, s_cz;
    __shared__ float s_val[32];
    __shared__ int   s_idx[32];
    __shared__ int   s_cur;
    if (tid == 0) s_cur = 0;
    __syncthreads();

    for (int it = 0; it < Ss; ++it) {
        int curi = s_cur;
        if (tid == 0) {
            float cx = x[curi * 3 + 0];
            float cy = x[curi * 3 + 1];
            float cz = x[curi * 3 + 2];
            s_cx = cx; s_cy = cy; s_cz = cz;
            float* o = new_xyz + ((size_t)b * Ss + it) * 3;
            o[0] = cx; o[1] = cy; o[2] = cz;
        }
        __syncthreads();
        float cx = s_cx, cy = s_cy, cz = s_cz;

        float bestv = -1.0f; int besti = 0;
#pragma unroll
        for (int i = 0; i < 16; ++i) {
            float dx = px[i] - cx, dy = py[i] - cy, dz = pz[i] - cz;
            float d = dx * dx + dy * dy + dz * dz;
            dist[i] = fminf(dist[i], d);
            if (dist[i] > bestv) { bestv = dist[i]; besti = i * 1024 + tid; }
        }
        // wave32 reduce
#pragma unroll
        for (int off = 16; off > 0; off >>= 1) {
            float ov = __shfl_down(bestv, off, 32);
            int   oi = __shfl_down(besti, off, 32);
            if (ov > bestv) { bestv = ov; besti = oi; }
        }
        if ((tid & 31) == 0) { s_val[tid >> 5] = bestv; s_idx[tid >> 5] = besti; }
        __syncthreads();
        if (tid < 32) {
            bestv = s_val[tid]; besti = s_idx[tid];
#pragma unroll
            for (int off = 16; off > 0; off >>= 1) {
                float ov = __shfl_down(bestv, off, 32);
                int   oi = __shfl_down(besti, off, 32);
                if (ov > bestv) { bestv = ov; besti = oi; }
            }
            if (tid == 0) s_cur = besti;
        }
        __syncthreads();
    }
}

// ---------------------------------------------------------------------------
// Kernel 2: ball query. One wave32 per centroid; in-order append of the first
// NSs indices within radius via ballot + prefix popcount; pad with first hit.
// ---------------------------------------------------------------------------
__global__ __launch_bounds__(128) void ballq_kernel(const float* __restrict__ xyz,
                                                    const float* __restrict__ new_xyz,
                                                    int* __restrict__ nidx) {
    const int lane = threadIdx.x & 31;
    const int gid  = blockIdx.x * 4 + (threadIdx.x >> 5);   // 0..B*S-1
    const int b    = gid >> 10;                              // S == 1024
    const float* x = xyz + (size_t)b * Nn * 3;
    const float* c = new_xyz + (size_t)gid * 3;
    const float cx = c[0], cy = c[1], cz = c[2];
    int* out = nidx + (size_t)gid * NSs;

    const float r2 = RADIUS * RADIUS;
    int found = 0, first = 0;
    for (int base = 0; base < Nn && found < NSs; base += 32) {
        int p = base + lane;
        float dx = x[p * 3 + 0] - cx;
        float dy = x[p * 3 + 1] - cy;
        float dz = x[p * 3 + 2] - cz;
        bool in = (dx * dx + dy * dy + dz * dz) <= r2;
        unsigned mask = (unsigned)__ballot(in);
        if (found == 0 && mask) first = base + __ffs(mask) - 1;
        int pre = __popc(mask & ((1u << lane) - 1u));
        if (in && (found + pre) < NSs) out[found + pre] = p;
        found += __popc(mask);
    }
    int fcount = found < NSs ? found : NSs;
    if (lane >= fcount && lane < NSs) out[lane] = first;
}

// ---------------------------------------------------------------------------
// WMMA fragment loaders. A-tile rows are row-major [m][k]; per-lane data is
// two contiguous 8-element K-chunks -> 2x ds_load_b128. Weights are stored
// TRANSPOSED in LDS ([n][k]) so a B fragment is one contiguous 16-element
// K-run per lane -> 2x ds_load_b128.
// ---------------------------------------------------------------------------
__device__ __forceinline__ v16h load_a_frag(const _Float16* Abuf, int ldk,
                                            int mt, int kt, int lane) {
    const int m    = mt * 16 + (lane & 15);
    const int half = lane >> 4;
    const _Float16* p = Abuf + m * ldk + kt * 32 + half * 16;
    v8h lo = *(const v8h*)(p);          // k = half*16 + 0..7
    v8h hi = *(const v8h*)(p + 8);      // k = half*16 + 8..15
    return __builtin_shufflevector(lo, hi, 0, 1, 2, 3, 4, 5, 6, 7,
                                           8, 9, 10, 11, 12, 13, 14, 15);
}

__device__ __forceinline__ v16h load_b_frag(const _Float16* Wt, int ldk,
                                            int kt, int nt, int lane) {
    const int n = nt * 16 + (lane & 15);
    const _Float16* p = Wt + n * ldk + kt * 32 + (lane >> 4) * 16;
    v8h lo = *(const v8h*)(p);          // k = half*16 + 0..7
    v8h hi = *(const v8h*)(p + 8);      // k = half*16 + 8..15
    return __builtin_shufflevector(lo, hi, 0, 1, 2, 3, 4, 5, 6, 7,
                                           8, 9, 10, 11, 12, 13, 14, 15);
}

__device__ __forceinline__ v8f wmma_f16(v16h a, v16h b, v8f c) {
    return __builtin_amdgcn_wmma_f32_16x16x32_f16(false, a, false, b,
                                                  (short)0, c, false, false);
}

// ---------------------------------------------------------------------------
// Kernel 3: fused gather + 3-layer MLP (WMMA f16, f32 accum) + max-pool.
// 4 waves per block, one centroid per wave. Weights shared in LDS as f16,
// transposed [out][in]. Input row layout: [feat(64) | xyz(3) | pad(29)],
// with W0's K rows permuted to match.
// ---------------------------------------------------------------------------
__global__ __launch_bounds__(128) void mlp_kernel(
    const float* __restrict__ xyz, const float* __restrict__ feat,
    const float* __restrict__ new_xyz, const int* __restrict__ nidx,
    const float* __restrict__ W0, const float* __restrict__ b0,
    const float* __restrict__ W1, const float* __restrict__ b1,
    const float* __restrict__ W2, const float* __restrict__ b2,
    float* __restrict__ new_feat) {

    __shared__ _Float16 sW0t[64 * K0P];    // [n=64][k'=96] (k'>=67 zero)
    __shared__ _Float16 sW1t[64 * 64];     // [n][k]
    __shared__ _Float16 sW2t[128 * 64];    // [n][k]
    __shared__ float    sB0[64], sB1[64], sB2[128];
    __shared__ _Float16 sA[4][32 * K0P];   // per-wave activation (stride 96)
    __shared__ _Float16 sH[4][32 * 128];   // per-wave activation (stride 128)

    const int tid  = threadIdx.x;
    const int lane = tid & 31;
    const int wv   = tid >> 5;
    const int gid  = blockIdx.x * 4 + wv;  // centroid id, 0..B*S-1
    const int b    = gid >> 10;

    // cooperative weight load, transpose, fp16 down-convert.
    // W0 K-permutation: k'<64 -> original row k'+3 (features); 64..66 -> xyz.
    for (int i = tid; i < 64 * K0P; i += 128) {
        int n = i / K0P, kp = i - n * K0P;
        float w = 0.0f;
        if (kp < 64)      w = W0[(kp + 3) * 64 + n];
        else if (kp < 67) w = W0[(kp - 64) * 64 + n];
        sW0t[i] = (_Float16)w;
    }
    for (int i = tid; i < 64 * 64; i += 128) {
        int n = i >> 6, k = i & 63;
        sW1t[i] = (_Float16)W1[k * 64 + n];
    }
    for (int i = tid; i < 128 * 64; i += 128) {
        int n = i >> 6, k = i & 63;
        sW2t[i] = (_Float16)W2[k * 128 + n];
    }
    if (tid < 64) { sB0[tid] = b0[tid]; sB1[tid] = b1[tid]; }
    if (tid < 128) sB2[tid] = b2[tid];

    // gather: one neighbor row per lane -> sA[wv] (32 x 96, layout feat|xyz|0)
    {
        const float* c = new_xyz + (size_t)gid * 3;
        float cx = c[0], cy = c[1], cz = c[2];
        int idx = nidx[(size_t)gid * NSs + lane];
        const float*  pxyz = xyz + ((size_t)b * Nn + idx) * 3;
        const float4* pf4  = (const float4*)(feat + ((size_t)b * Nn + idx) * Cc);
        __builtin_prefetch(pf4, 0, 0);
        _Float16* row = &sA[wv][lane * K0P];
#pragma unroll
        for (int j = 0; j < 8; ++j) {            // 64 features, 8 halfs/store
            float4 f0 = pf4[2 * j];
            float4 f1 = pf4[2 * j + 1];
            v8h h;
            h[0] = (_Float16)f0.x; h[1] = (_Float16)f0.y;
            h[2] = (_Float16)f0.z; h[3] = (_Float16)f0.w;
            h[4] = (_Float16)f1.x; h[5] = (_Float16)f1.y;
            h[6] = (_Float16)f1.z; h[7] = (_Float16)f1.w;
            *(v8h*)(row + 8 * j) = h;
        }
        row[64] = (_Float16)(pxyz[0] - cx);
        row[65] = (_Float16)(pxyz[1] - cy);
        row[66] = (_Float16)(pxyz[2] - cz);
        row[67] = (_Float16)0.0f;
        row[68] = (_Float16)0.0f;
        row[69] = (_Float16)0.0f;
        row[70] = (_Float16)0.0f;
        row[71] = (_Float16)0.0f;
        v8h z = {};
        *(v8h*)(row + 72) = z;
        *(v8h*)(row + 80) = z;
        *(v8h*)(row + 88) = z;
    }
    __syncthreads();

    // ---- layer 0: (32x96) @ (96x64) -> sH (stride 128), ReLU ----
#pragma unroll
    for (int mt = 0; mt < 2; ++mt) {
#pragma unroll
        for (int nt = 0; nt < 4; ++nt) {
            v8f acc = {};
#pragma unroll
            for (int kt = 0; kt < 3; ++kt) {
                v16h a  = load_a_frag(sA[wv], K0P, mt, kt, lane);
                v16h bf = load_b_frag(sW0t, K0P, kt, nt, lane);
                acc = wmma_f16(a, bf, acc);
            }
            int n = nt * 16 + (lane & 15);
            float bias = sB0[n];
#pragma unroll
            for (int r = 0; r < 8; ++r) {
                int m = mt * 16 + (lane >> 4) * 8 + r;
                float v = acc[r] + bias;
                sH[wv][m * 128 + n] = (_Float16)(v > 0.0f ? v : 0.0f);
            }
        }
    }
    __syncthreads();

    // ---- layer 1: (32x64) @ (64x64) -> sA (stride 96), ReLU ----
#pragma unroll
    for (int mt = 0; mt < 2; ++mt) {
#pragma unroll
        for (int nt = 0; nt < 4; ++nt) {
            v8f acc = {};
#pragma unroll
            for (int kt = 0; kt < 2; ++kt) {
                v16h a  = load_a_frag(sH[wv], 128, mt, kt, lane);
                v16h bf = load_b_frag(sW1t, 64, kt, nt, lane);
                acc = wmma_f16(a, bf, acc);
            }
            int n = nt * 16 + (lane & 15);
            float bias = sB1[n];
#pragma unroll
            for (int r = 0; r < 8; ++r) {
                int m = mt * 16 + (lane >> 4) * 8 + r;
                float v = acc[r] + bias;
                sA[wv][m * K0P + n] = (_Float16)(v > 0.0f ? v : 0.0f);
            }
        }
    }
    __syncthreads();

    // ---- layer 2: (32x64) @ (64x128), ReLU, fused max-pool over 32 rows ----
#pragma unroll
    for (int nt = 0; nt < 8; ++nt) {
        float colmax = -3.0e38f;
#pragma unroll
        for (int mt = 0; mt < 2; ++mt) {
            v8f acc = {};
#pragma unroll
            for (int kt = 0; kt < 2; ++kt) {
                v16h a  = load_a_frag(sA[wv], K0P, mt, kt, lane);
                v16h bf = load_b_frag(sW2t, 64, kt, nt, lane);
                acc = wmma_f16(a, bf, acc);
            }
            int n = nt * 16 + (lane & 15);
            float bias = sB2[n];
#pragma unroll
            for (int r = 0; r < 8; ++r) {
                float v = acc[r] + bias;
                v = v > 0.0f ? v : 0.0f;
                colmax = fmaxf(colmax, v);
            }
        }
        // lanes L and L^16 hold the two 8-row halves of the same column n
        colmax = fmaxf(colmax, __shfl_xor(colmax, 16, 32));
        if (lane < 16)
            new_feat[(size_t)gid * H3 + nt * 16 + lane] = colmax;
    }
}

// ---------------------------------------------------------------------------
extern "C" void kernel_launch(void* const* d_in, const int* in_sizes, int n_in,
                              void* d_out, int out_size, void* d_ws, size_t ws_size,
                              hipStream_t stream) {
    const float* xyz  = (const float*)d_in[0];
    const float* feat = (const float*)d_in[1];
    const float* W0   = (const float*)d_in[2];
    const float* b0   = (const float*)d_in[3];
    const float* W1   = (const float*)d_in[4];
    const float* b1   = (const float*)d_in[5];
    const float* W2   = (const float*)d_in[6];
    const float* b2   = (const float*)d_in[7];

    float* out      = (float*)d_out;
    float* new_xyz  = out;                       // (B,S,3)
    float* new_feat = out + (size_t)Bb * Ss * 3; // (B,S,128)
    int*   nidx     = (int*)d_ws;                // (B,S,NS) = 512 KB scratch

    fps_kernel  <<<Bb,          1024, 0, stream>>>(xyz, new_xyz);
    ballq_kernel<<<Bb * Ss / 4, 128,  0, stream>>>(xyz, new_xyz, nidx);
    mlp_kernel  <<<Bb * Ss / 4, 128,  0, stream>>>(xyz, feat, new_xyz, nidx,
                                                   W0, b0, W1, b1, W2, b2,
                                                   new_feat);
}
